// SelfAttention_3100966388030
// MI455X (gfx1250) — matrix operand order repaired and checked by reading, not compiled, not run on hardware
//
#include <hip/hip_runtime.h>

// ---------------------------------------------------------------------------
// MI455X self-attention: f16 WMMA everywhere, fp32 accumulate.
// Working set (Qh/Kh/Vt/Aout f16 + Wo16) ~= 67MB -> fully L2 resident (192MB),
// so WMMA fragments are loaded straight from global (L2) without LDS staging,
// except the P-tile C->A relayout which round-trips 1KB of wave-private LDS.
// ---------------------------------------------------------------------------

typedef __attribute__((ext_vector_type(16))) _Float16 v16h;
typedef __attribute__((ext_vector_type(8)))  _Float16 v8h;
typedef __attribute__((ext_vector_type(8)))  float    v8f;
typedef __attribute__((ext_vector_type(4)))  float    v4f;

#define WMMA_F16F32(a, b, c)                                                   \
  __builtin_amdgcn_wmma_f32_16x16x32_f16(false, (a), false, (b), (short)0,     \
                                         (c), false, false)

#define NBATCH   8
#define SEQ      1024
#define EMBED    1024
#define HEADS    16
#define HEAD_DIM 64
#define INV_SQRT_EMBED 0.03125f  // 1/sqrt(1024)

// --- A-fragment (16xK tile, row-major f16 source, ld halves per row) --------
// a[0..7]  = row(lane&15), K = koff + hi*8  + 0..7
// a[8..15] = row(lane&15), K = koff + hi*8  + 16..23   (hi = lane>=16)
__device__ __forceinline__ v16h load_a_f16(const _Float16* tile, int ld,
                                           int lane, int koff) {
  int m  = lane & 15;
  int hi = (lane >> 4) & 1;
  const _Float16* p = tile + (size_t)m * ld + koff + hi * 8;
  union { v16h v; v8h h[2]; } u;
  u.h[0] = *(const v8h*)(p);
  u.h[1] = *(const v8h*)(p + 16);
  return u.v;
}

// Same A layout but from fp32 source (converted on the fly).
__device__ __forceinline__ v16h load_a_f32(const float* tile, int ld,
                                           int lane, int koff) {
  int m  = lane & 15;
  int hi = (lane >> 4) & 1;
  const float* p = tile + (size_t)m * ld + koff + hi * 8;
  v4f x0 = *(const v4f*)(p);
  v4f x1 = *(const v4f*)(p + 4);
  v4f y0 = *(const v4f*)(p + 16);
  v4f y1 = *(const v4f*)(p + 20);
  v16h a;
#pragma unroll
  for (int i = 0; i < 4; ++i) {
    a[i]      = (_Float16)x0[i];
    a[4 + i]  = (_Float16)x1[i];
    a[8 + i]  = (_Float16)y0[i];
    a[12 + i] = (_Float16)y1[i];
  }
  return a;
}

// --- B-fragment (Kx16 tile). Source "bt" stores B columns contiguously:
// bt row n (= output col) holds B[k][n] for consecutive k.
// b[0..7] = B[koff + hi*8 + 0..7][lane&15], b[8..15] = B[koff + hi*8 + 16..23][lane&15].
__device__ __forceinline__ v16h load_bT(const _Float16* bt, int ld,
                                        int lane, int koff) {
  int nn = lane & 15;
  int hi = (lane >> 4) & 1;
  const _Float16* p = bt + (size_t)nn * ld + koff + hi * 8;
  union { v16h v; v8h h[2]; } u;
  u.h[0] = *(const v8h*)(p);
  u.h[1] = *(const v8h*)(p + 16);
  return u.v;
}

// 16-lane butterfly reductions (rows of the C layout live in 16-lane halves).
__device__ __forceinline__ float rowmax16(float x) {
#pragma unroll
  for (int m = 1; m < 16; m <<= 1) x = fmaxf(x, __shfl_xor(x, m, 32));
  return x;
}
__device__ __forceinline__ float rowsum16(float x) {
#pragma unroll
  for (int m = 1; m < 16; m <<= 1) x += __shfl_xor(x, m, 32);
  return x;
}

// ---------------------------------------------------------------------------
// Kernel 1: per-head projections. 16 consecutive rows of X = the 16 heads of
// one token, all sharing the same 64x64 weight -> perfect 16x64x64 WMMA tile.
// blockIdx.y selects {q,k,v}. V output is transposed to [n][h][d][l].
// Both layouts share the h coefficient (h<<16 halves), so the store epilogue
// is a single branch-free address: n<<20 | h<<16 | e*estride + l*lstride.
// ---------------------------------------------------------------------------
__global__ __launch_bounds__(256) void proj_kernel(
    const float* __restrict__ q, const float* __restrict__ k,
    const float* __restrict__ v, const float* __restrict__ Wq,
    const float* __restrict__ Wk, const float* __restrict__ Wv,
    _Float16* __restrict__ Qh, _Float16* __restrict__ Kh,
    _Float16* __restrict__ Vt) {
  __shared__ _Float16 w16[64 * 64];  // W row-major [e][d]; B col e = W row e
  const int which = blockIdx.y;
  const float* X = which == 0 ? q : (which == 1 ? k : v);
  const float* W = which == 0 ? Wq : (which == 1 ? Wk : Wv);
  _Float16* O    = which == 0 ? Qh : (which == 1 ? Kh : Vt);
  const int estride = (which == 2) ? SEQ : 1;   // e coeff (halves)
  const int lstride = (which == 2) ? 1 : HEAD_DIM;  // l coeff (halves)

  for (int i = threadIdx.x; i < 64 * 64; i += 256) w16[i] = (_Float16)W[i];
  __syncthreads();

  const int lane = threadIdx.x & 31;
  const int wave = threadIdx.x >> 5;
  const int token = blockIdx.x * 8 + wave;  // tile == token (16 heads)
  const int n = token >> 10;
  const int l = token & 1023;
  const float* Arow = X + (size_t)token * 16 * 64;  // 16 rows x 64

  v16h a0 = load_a_f32(Arow, 64, lane, 0);
  v16h a1 = load_a_f32(Arow, 64, lane, 32);

  const int hi = (lane >> 4) & 1;
  const int nn = lane & 15;
  // Per-lane invariant part of the output address (h = r + hi*8 added below).
  _Float16* Olane = O + ((size_t)n << 20) + ((size_t)(hi * 8) << 16) +
                    (size_t)l * lstride;
#pragma unroll
  for (int nt = 0; nt < 4; ++nt) {
    v8f acc = {};
    v16h b0 = load_bT(&w16[nt * 16 * 64], 64, lane, 0);
    v16h b1 = load_bT(&w16[nt * 16 * 64], 64, lane, 32);
    acc = WMMA_F16F32(a0, b0, acc);
    acc = WMMA_F16F32(a1, b1, acc);
    const int e = nt * 16 + nn;
    _Float16* Obase = Olane + (size_t)e * estride;
#pragma unroll
    for (int r = 0; r < 8; ++r)  // r<<16 halves folds into store imm offset
      Obase[(size_t)r << 16] = (_Float16)acc[r];
  }
}

// ---------------------------------------------------------------------------
// Kernel 2: flash attention. One wave owns 16 query rows; loops over 32-key
// chunks: S = Q·K^T (4 WMMA), mask BEFORE 1/32 scale (as reference), online
// softmax with shuffle reductions, P relayout via LDS, O += P·V (4 WMMA).
// ---------------------------------------------------------------------------
__global__ __launch_bounds__(256) void attn_kernel(
    const _Float16* __restrict__ Qh, const _Float16* __restrict__ Kh,
    const _Float16* __restrict__ Vt, const int* __restrict__ mask,
    _Float16* __restrict__ Aout) {
  __shared__ _Float16 plds[8][16 * 32];  // per-wave 16x32 P tile
  const int lane = threadIdx.x & 31;
  const int wave = threadIdx.x >> 5;
  const int nh = blockIdx.y;        // 0..127
  const int n = nh >> 4, h = nh & 15;
  const int qbase = blockIdx.x * 128 + wave * 16;

  const _Float16* Qp = Qh + (size_t)nh * SEQ * HEAD_DIM;
  const _Float16* Kp = Kh + (size_t)nh * SEQ * HEAD_DIM;
  const _Float16* Vp = Vt + (size_t)nh * HEAD_DIM * SEQ;
  const int* Mp = mask + ((size_t)n * SEQ + qbase) * SEQ;

  const v16h qa0 = load_a_f16(Qp + (size_t)qbase * HEAD_DIM, HEAD_DIM, lane, 0);
  const v16h qa1 = load_a_f16(Qp + (size_t)qbase * HEAD_DIM, HEAD_DIM, lane, 32);

  float mrow[8], lrow[8], alpha[8];
  v8f acc[4] = {v8f{}, v8f{}, v8f{}, v8f{}};
#pragma unroll
  for (int r = 0; r < 8; ++r) { mrow[r] = -3.0e38f; lrow[r] = 0.f; }

  const int hi = (lane >> 4) & 1;
  const int nn = lane & 15;
  _Float16* pw = &plds[wave][0];

  for (int kt = 0; kt < SEQ / 32; ++kt) {
    const int kb = kt * 32;
    if (kt + 1 < SEQ / 32)  // speculative prefetch of next K tile into caches
      __builtin_prefetch(Kp + (size_t)(kb + 32 + lane) * HEAD_DIM, 0, 0);

    // S tiles: columns kb+0..15 and kb+16..31; K rows are B columns.
    v8f s0 = {}, s1 = {};
    s0 = WMMA_F16F32(qa0, load_bT(Kp + (size_t)kb * HEAD_DIM, HEAD_DIM, lane, 0), s0);
    s0 = WMMA_F16F32(qa1, load_bT(Kp + (size_t)kb * HEAD_DIM, HEAD_DIM, lane, 32), s0);
    s1 = WMMA_F16F32(qa0, load_bT(Kp + (size_t)(kb + 16) * HEAD_DIM, HEAD_DIM, lane, 0), s1);
    s1 = WMMA_F16F32(qa1, load_bT(Kp + (size_t)(kb + 16) * HEAD_DIM, HEAD_DIM, lane, 32), s1);

#pragma unroll
    for (int r = 0; r < 8; ++r) {
      const int qm = r + hi * 8;  // C-row -> query row within strip
      const int m0 = Mp[(size_t)qm * SEQ + kb + nn];
      const int m1 = Mp[(size_t)qm * SEQ + kb + 16 + nn];
      float x0 = (m0 == 0) ? -1.0e20f : s0[r];
      float x1 = (m1 == 0) ? -1.0e20f : s1[r];
      x0 *= INV_SQRT_EMBED;
      x1 *= INV_SQRT_EMBED;
      const float tmax = rowmax16(fmaxf(x0, x1));
      const float mnew = fmaxf(mrow[r], tmax);
      alpha[r] = __expf(mrow[r] - mnew);
      const float p0 = __expf(x0 - mnew);
      const float p1 = __expf(x1 - mnew);
      lrow[r] = lrow[r] * alpha[r] + rowsum16(p0 + p1);
      mrow[r] = mnew;
      pw[qm * 32 + nn]      = (_Float16)p0;  // C layout -> row-major LDS
      pw[qm * 32 + 16 + nn] = (_Float16)p1;
    }
    asm volatile("s_wait_dscnt 0x0" ::: "memory");  // stores visible to loads

#pragma unroll
    for (int dt = 0; dt < 4; ++dt)
#pragma unroll
      for (int r = 0; r < 8; ++r) acc[dt][r] *= alpha[r];

    const v16h pa = load_a_f16(pw, 32, lane, 0);  // P as 16x32 A fragment
#pragma unroll
    for (int dt = 0; dt < 4; ++dt) {
      v16h vb = load_bT(Vp + (size_t)(dt * 16) * SEQ, SEQ, lane, kb);
      acc[dt] = WMMA_F16F32(pa, vb, acc[dt]);
    }
  }

#pragma unroll
  for (int r = 0; r < 8; ++r) lrow[r] = 1.0f / lrow[r];
  _Float16* Abase = Aout + ((size_t)n * SEQ + qbase + hi * 8) * EMBED +
                    h * HEAD_DIM + nn;
#pragma unroll
  for (int dt = 0; dt < 4; ++dt) {
#pragma unroll
    for (int r = 0; r < 8; ++r)
      Abase[(size_t)r * EMBED + dt * 16] = (_Float16)(acc[dt][r] * lrow[r]);
  }
}

// ---------------------------------------------------------------------------
// Kernel 3a: Wo -> f16 once (2MB, stays in L2 for the GEMM).
// ---------------------------------------------------------------------------
__global__ __launch_bounds__(256) void cvt_wo(const float* __restrict__ Wo,
                                              _Float16* __restrict__ Wo16) {
  const size_t i = (size_t)blockIdx.x * 256 + threadIdx.x;
  Wo16[i] = (_Float16)Wo[i];
}

// ---------------------------------------------------------------------------
// Kernel 3b: Y = Aout @ Wo^T + bo. One wave: 16 rows x 64 cols, 128 WMMAs.
// Wo row e == B column e (contiguous), so load_bT reads straight from Wo16.
// ---------------------------------------------------------------------------
__global__ __launch_bounds__(256) void outproj_kernel(
    const _Float16* __restrict__ Aout, const _Float16* __restrict__ Wo16,
    const float* __restrict__ bo, float* __restrict__ Y) {
  const int lane = threadIdx.x & 31;
  const int wave = threadIdx.x >> 5;
  const int mbase = (blockIdx.x * 8 + wave) * 16;  // 8192 rows total
  const int cbase = blockIdx.y * 64;               // 1024 cols total

  v8f acc[4] = {v8f{}, v8f{}, v8f{}, v8f{}};
  const _Float16* Atile = Aout + (size_t)mbase * EMBED;
  for (int kc = 0; kc < EMBED / 32; ++kc) {
    const v16h a = load_a_f16(Atile, EMBED, lane, kc * 32);
#pragma unroll
    for (int nt = 0; nt < 4; ++nt) {
      v16h b = load_bT(Wo16 + (size_t)(cbase + nt * 16) * EMBED, EMBED, lane,
                       kc * 32);
      acc[nt] = WMMA_F16F32(a, b, acc[nt]);
    }
  }

  const int hi = (lane >> 4) & 1;
  const int nn = lane & 15;
  float* Ybase = Y + (size_t)(mbase + hi * 8) * EMBED + cbase + nn;
#pragma unroll
  for (int nt = 0; nt < 4; ++nt) {
    const float bias = bo[cbase + nt * 16 + nn];
#pragma unroll
    for (int r = 0; r < 8; ++r)
      Ybase[(size_t)r * EMBED + nt * 16] = acc[nt][r] + bias;
  }
}

// ---------------------------------------------------------------------------
extern "C" void kernel_launch(void* const* d_in, const int* in_sizes, int n_in,
                              void* d_out, int out_size, void* d_ws,
                              size_t ws_size, hipStream_t stream) {
  const float* values = (const float*)d_in[0];
  const float* keys   = (const float*)d_in[1];
  const float* query  = (const float*)d_in[2];
  const int*   mask   = (const int*)d_in[3];
  const float* Wv     = (const float*)d_in[4];
  const float* Wk     = (const float*)d_in[5];
  const float* Wq     = (const float*)d_in[6];
  const float* Wo     = (const float*)d_in[7];
  const float* bo     = (const float*)d_in[8];
  float* Y = (float*)d_out;

  const size_t HALVES = (size_t)NBATCH * SEQ * EMBED;  // 8M halves per tensor
  _Float16* ws   = (_Float16*)d_ws;
  _Float16* Qh   = ws;
  _Float16* Kh   = Qh + HALVES;
  _Float16* Vt   = Kh + HALVES;
  _Float16* Aout = Vt + HALVES;
  _Float16* Wo16 = Aout + HALVES;  // + 1M halves -> total ~67MB of ws

  proj_kernel<<<dim3(1024, 3), 256, 0, stream>>>(query, keys, values, Wq, Wk,
                                                 Wv, Qh, Kh, Vt);
  cvt_wo<<<dim3(4096), 256, 0, stream>>>(Wo, Wo16);
  attn_kernel<<<dim3(8, 128), 256, 0, stream>>>(Qh, Kh, Vt, mask, Aout);
  outproj_kernel<<<dim3(64, 16), 256, 0, stream>>>(Aout, Wo16, bo, Y);
}
